// MultiHeadAttention_68212670595179
// MI455X (gfx1250) — matrix operand order repaired and checked by reading, not compiled, not run on hardware
//
#include <hip/hip_runtime.h>

typedef __attribute__((ext_vector_type(16))) __bf16 v16bf;
typedef __attribute__((ext_vector_type(8)))  float  v8f;
typedef __attribute__((ext_vector_type(4)))  unsigned int u32x4;
typedef __attribute__((ext_vector_type(8)))  int i32x8;
typedef __attribute__((ext_vector_type(4)))  int i32x4;

#define GM 16384   // B*S rows
#define GN 1280    // output dim
#define GK 1280    // inner dim

static __device__ __forceinline__ unsigned short f2bf(float f) {
  unsigned int u = __float_as_uint(f);
  u += 0x7FFFu + ((u >> 16) & 1u);          // round-to-nearest-even
  return (unsigned short)(u >> 16);
}
static __device__ __forceinline__ float bf2f(unsigned short h) {
  return __uint_as_float(((unsigned int)h) << 16);
}

// ---------------- fp32 -> bf16 bulk convert (for x) ----------------
__global__ void cvt_f32_bf16_x4(const float* __restrict__ src,
                                unsigned short* __restrict__ dst, int n4) {
  int i = blockIdx.x * blockDim.x + threadIdx.x;
  if (i >= n4) return;
  float4 f = ((const float4*)src)[i];
  uint2 o;
  o.x = (unsigned)f2bf(f.x) | ((unsigned)f2bf(f.y) << 16);
  o.y = (unsigned)f2bf(f.z) | ((unsigned)f2bf(f.w) << 16);
  ((uint2*)dst)[i] = o;
}

// ------- weight transpose+convert: W[k][n] f32 -> Wt[n][k] bf16 -------
__global__ void wtrans(const float* __restrict__ W,
                       unsigned short* __restrict__ Wt) {
  __shared__ float tile[32][33];
  int n0 = blockIdx.x * 32, k0 = blockIdx.y * 32;
  int tx = threadIdx.x, ty = threadIdx.y;
  for (int i = 0; i < 4; ++i)
    tile[ty + i * 8][tx] = W[(size_t)(k0 + ty + i * 8) * GN + n0 + tx];
  __syncthreads();
  for (int i = 0; i < 4; ++i)
    Wt[(size_t)(n0 + ty + i * 8) * GK + k0 + tx] = f2bf(tile[tx][ty + i * 8]);
}

// ---------------- TDM: stage a 128x32 bf16 tile (row stride GK) into LDS ----------------
// D# per CDNA5 ISA §8.3/8.4. LDS pad: 16 DWORDs data + 8 DWORDs pad -> 48-element rows.
static __device__ __forceinline__ void tdm_load_tile(unsigned lds_addr,
                                                     unsigned long long gaddr,
                                                     int rows_total) {
  u32x4 g0;
  g0.x = 1u;                                                   // count=1 (valid D#)
  g0.y = lds_addr;                                             // lds_addr [63:32]
  g0.z = (unsigned)gaddr;                                      // global_addr [95:64]
  g0.w = (unsigned)((gaddr >> 32) & 0x01FFFFFFu) | (2u << 30); // addr[56:32] | type=2
  i32x8 g1;
  g1.s0 = (int)0x0ED10000u;        // data_size=2B | pad_enable | interval=16DW | amount=8DW
  g1.s1 = (int)(GK << 16);         // tensor_dim0 = GK  (low16 at bits[63:48])
  g1.s2 = (int)(rows_total << 16); // tensor_dim1 low16 at bits[95:80]
  g1.s3 = (int)(32 << 16);         // tile_dim0 = 32 elements (bits[127:112])
  g1.s4 = 128;                     // tile_dim1 = 128 rows
  g1.s5 = GK;                      // tensor_dim0_stride = GK elements
  g1.s6 = 0;
  g1.s7 = 0;
  i32x4 z4 = {0, 0, 0, 0};
  i32x8 z8 = {0, 0, 0, 0, 0, 0, 0, 0};
  __builtin_amdgcn_tensor_load_to_lds(g0, g1, z4, z4, z8, 0);
}

// ---------------- WMMA bf16 GEMM:  C = A[GM,GK] * Wt[GN,GK]^T + bias ----------------
union Frag { v16bf v; uint4 u[2]; };

template <int OUT_BF16>
__global__ __launch_bounds__(256) void gemm_bf16(
    const unsigned short* __restrict__ A,    // [GM][GK]   bf16, row-major
    const unsigned short* __restrict__ Bm,   // [GN][GK]   bf16 (pre-transposed weight)
    const float* __restrict__ bias,          // [GN]
    unsigned short* __restrict__ outb,       // bf16 out (OUT_BF16=1)
    float* __restrict__ outf)                // f32  out (OUT_BF16=0)
{
  __shared__ unsigned short As[2][128 * 48]; // rows padded 32 -> 48 by TDM (96B stride)
  __shared__ unsigned short Bs[2][128 * 48];

  const int t  = threadIdx.x;
  const int m0 = blockIdx.y * 128;
  const int n0 = blockIdx.x * 128;
  const int wv = t >> 5, lane = t & 31;
  const int hf = lane >> 4, lr = lane & 15;  // half-wave select, lane-in-half
  const int wm = wv & 3, wn = wv >> 2;       // wave tile: rows wm*32.., cols wn*64..

  const unsigned long long gA = (unsigned long long)(uintptr_t)A  + (size_t)m0 * GK * 2;
  const unsigned long long gB = (unsigned long long)(uintptr_t)Bm + (size_t)n0 * GK * 2;
  const unsigned ldsA[2] = { (unsigned)(uintptr_t)&As[0][0], (unsigned)(uintptr_t)&As[1][0] };
  const unsigned ldsB[2] = { (unsigned)(uintptr_t)&Bs[0][0], (unsigned)(uintptr_t)&Bs[1][0] };

  v8f acc[2][4] = {};

  const int NIT = GK / 32;                   // 40 K-steps
  // prologue: wave0 stages A tile 0, wave1 stages B tile 0 (parallel TDM streams)
  if (wv == 0) tdm_load_tile(ldsA[0], gA, GM);
  if (wv == 1) tdm_load_tile(ldsB[0], gB, GN);

  for (int it = 0; it < NIT; ++it) {
    const int buf = it & 1;
    if (it + 1 < NIT) {                      // issue next tile into other buffer
      const unsigned long long koff = (unsigned long long)(it + 1) * 64ull; // 32 elems * 2B
      if (wv == 0) tdm_load_tile(ldsA[buf ^ 1], gA + koff, GM);
      if (wv == 1) tdm_load_tile(ldsB[buf ^ 1], gB + koff, GN);
      if (wv < 2) __builtin_amdgcn_s_wait_tensorcnt((short)1);  // current tile landed
    } else {
      if (wv < 2) __builtin_amdgcn_s_wait_tensorcnt((short)0);
    }
    __syncthreads();

    const unsigned short* as = As[buf];
    const unsigned short* bs = Bs[buf];
    // ---- fragments per ISA 7.12.2 16-bit layouts ----
    Frag a[2], b[4];
    for (int mt = 0; mt < 2; ++mt) {
      int row = wm * 32 + mt * 16 + lr;      // A: lane lr -> M ; K = hf*8 ∪ 16+hf*8
      a[mt].u[0] = *(const uint4*)&as[row * 48 + hf * 8];
      a[mt].u[1] = *(const uint4*)&as[row * 48 + 16 + hf * 8];
    }
    for (int nt = 0; nt < 4; ++nt) {
      int n = wn * 64 + nt * 16 + lr;        // B: lane lr -> N ; K = hf*8 ∪ 16+hf*8
      b[nt].u[0] = *(const uint4*)&bs[n * 48 + hf * 8];
      b[nt].u[1] = *(const uint4*)&bs[n * 48 + 16 + hf * 8];
    }
    for (int mt = 0; mt < 2; ++mt)
      for (int nt = 0; nt < 4; ++nt)
        acc[mt][nt] = __builtin_amdgcn_wmma_f32_16x16x32_bf16(
            false, a[mt].v, false, b[nt].v, (short)0, acc[mt][nt], false, false);
    __syncthreads();                         // readers done -> buffer reusable
  }

  // ---- epilogue: C layout VGPR r, lane -> (M = r + 8*hf, N = lr) ----
  for (int nt = 0; nt < 4; ++nt) {
    int col = n0 + wn * 64 + nt * 16 + lr;
    float bval = bias[col];
    for (int mt = 0; mt < 2; ++mt)
      for (int r = 0; r < 8; ++r) {
        int row = m0 + wm * 32 + mt * 16 + r + hf * 8;
        float val = acc[mt][nt][r] + bval;
        if (OUT_BF16) outb[(size_t)row * GN + col] = f2bf(val);
        else          outf[(size_t)row * GN + col] = val;
      }
  }
}

// ---------------- per-position head-mixing attention ----------------
// scores[h,t] = (q[h,:] . k[t,:]) / sqrt(80); softmax over t; out[h,d] = sum_t p*v[t,d]
__global__ __launch_bounds__(256) void attn_softmax(
    const unsigned short* __restrict__ q,
    const unsigned short* __restrict__ k,
    const unsigned short* __restrict__ v,
    const float* __restrict__ pds,
    unsigned short* __restrict__ out)
{
  __shared__ float sc[8][16][17];
  const int w = threadIdx.x >> 5, lane = threadIdx.x & 31;
  const int p = blockIdx.x * 8 + w;                 // one wave per position
  const unsigned short* qp = q + (size_t)p * 1280;  // [h*80+d]
  const unsigned short* kp = k + (size_t)p * 1280;
  const unsigned short* vp = v + (size_t)p * 1280;

  const float scale = 0.11180339887498948f;         // 1/sqrt(80)
  for (int i = 0; i < 8; ++i) {                     // 256 (h,t) pairs / 32 lanes
    int idx = i * 32 + lane;
    int h = idx >> 4, tt = idx & 15;
    float s = 0.f;
    for (int d = 0; d < 80; ++d)
      s += bf2f(qp[h * 80 + d]) * bf2f(kp[tt * 80 + d]);
    sc[w][h][tt] = s * scale;
  }
  __syncthreads();
  if (lane < 16) {                                  // one row per lane
    float m = -3.4e38f;
    for (int tt = 0; tt < 16; ++tt) m = fmaxf(m, sc[w][lane][tt]);
    float e[16], sum = 0.f;
    for (int tt = 0; tt < 16; ++tt) { e[tt] = __expf(sc[w][lane][tt] - m); sum += e[tt]; }
    float inv = 1.f / sum;
    for (int tt = 0; tt < 16; ++tt) sc[w][lane][tt] = e[tt] * inv;
  }
  __syncthreads();
  for (int j = 0; j < 40; ++j) {                    // 1280 outputs / 32 lanes
    int idx = lane * 40 + j;                        // == h*80 + d
    int h = idx / 80, d = idx % 80;
    float o = 0.f;
    for (int tt = 0; tt < 16; ++tt)
      o += sc[w][h][tt] * bf2f(vp[tt * 80 + d]);
    out[(size_t)p * 1280 + idx] = f2bf(o * pds[d]);
  }
}

extern "C" void kernel_launch(void* const* d_in, const int* in_sizes, int n_in,
                              void* d_out, int out_size, void* d_ws, size_t ws_size,
                              hipStream_t stream) {
  const float* x   = (const float*)d_in[0];
  const float* Wq  = (const float*)d_in[1];
  const float* bq  = (const float*)d_in[2];
  const float* Wk  = (const float*)d_in[3];
  const float* bk  = (const float*)d_in[4];
  const float* Wv  = (const float*)d_in[5];
  const float* bv  = (const float*)d_in[6];
  const float* Wp  = (const float*)d_in[7];
  const float* bp  = (const float*)d_in[8];
  const float* pds = (const float*)d_in[9];

  // workspace layout (bytes): x_bf16 | 4 transposed weights bf16 | q,k,v bf16 | attn_out bf16
  char* ws = (char*)d_ws;
  const size_t xbytes = (size_t)GM * GK * 2;        //  41,943,040
  const size_t wel    = (size_t)GK * GN;            // elements per weight
  unsigned short* xb = (unsigned short*)ws;
  unsigned short* wt = (unsigned short*)(ws + xbytes);
  unsigned short* qb = (unsigned short*)(ws + xbytes + 4 * wel * 2);
  unsigned short* kb = qb + (size_t)GM * GN;
  unsigned short* vb = kb + (size_t)GM * GN;
  unsigned short* ao = vb + (size_t)GM * GN;

  // 1) convert inputs to bf16 (weights transposed for WMMA B-fragment loads)
  cvt_f32_bf16_x4<<<(GM * GK / 4 + 255) / 256, 256, 0, stream>>>(x, xb, GM * GK / 4);
  wtrans<<<dim3(40, 40), dim3(32, 8), 0, stream>>>(Wq, wt + 0 * wel);
  wtrans<<<dim3(40, 40), dim3(32, 8), 0, stream>>>(Wk, wt + 1 * wel);
  wtrans<<<dim3(40, 40), dim3(32, 8), 0, stream>>>(Wv, wt + 2 * wel);
  wtrans<<<dim3(40, 40), dim3(32, 8), 0, stream>>>(Wp, wt + 3 * wel);

  // 2) q/k/v projections (WMMA bf16 + TDM staging, bf16 outputs)
  dim3 gg(GN / 128, GM / 128);
  gemm_bf16<1><<<gg, 256, 0, stream>>>(xb, wt + 0 * wel, bq, qb, nullptr);
  gemm_bf16<1><<<gg, 256, 0, stream>>>(xb, wt + 1 * wel, bk, kb, nullptr);
  gemm_bf16<1><<<gg, 256, 0, stream>>>(xb, wt + 2 * wel, bv, vb, nullptr);

  // 3) per-position head-mixing attention + per_dim_scale
  attn_softmax<<<GM / 8, 256, 0, stream>>>(qb, kb, vb, pds, ao);

  // 4) output projection (WMMA bf16 + TDM staging, f32 output + bias into d_out)
  gemm_bf16<0><<<gg, 256, 0, stream>>>(ao, wt + 3 * wel, bp, nullptr, (float*)d_out);
}
